// selfattention_75265006895620
// MI455X (gfx1250) — matrix operand order repaired and checked by reading, not compiled, not run on hardware
//
#include <hip/hip_runtime.h>
#include <cstdint>
#include <cstddef>

#define DEVFN static __device__ __forceinline__

typedef __bf16 bf16;
typedef __attribute__((ext_vector_type(16))) __bf16 v16bf;
typedef __attribute__((ext_vector_type(8)))  __bf16 v8bf;
typedef __attribute__((ext_vector_type(4)))  __bf16 v4bf;
typedef __attribute__((ext_vector_type(8)))  float  v8f;
typedef __attribute__((ext_vector_type(4)))  float  v4f;

constexpr int kB  = 8;
constexpr int kN  = 4096;
constexpr int kV  = 1024;
constexpr int kKQ = 256;
constexpr int TLD = 40;                 // LDS tile leading dim (80B rows, 16B aligned)
constexpr int TILE_A = 256 * TLD;       // A tile: 256 rows x 32 K
constexpr int TILE_B = 128 * TLD;       // B tile: 128 cols x 32 K

// ---------------------------------------------------------------------------
// WMMA helpers (CDNA5 wave32, 16x16x32 bf16 -> f32)
// ---------------------------------------------------------------------------
DEVFN v8f wmma_bf16(v16bf a, v16bf b, v8f c) {
  return __builtin_amdgcn_wmma_f32_16x16x32_bf16(false, a, false, b, (short)0, c,
                                                 false, false);
}

// A fragment: 16x32 (MxK) row-major, leading dim ld.
DEVFN v16bf load_a_frag(const bf16* p, int ld) {
  const int lane = threadIdx.x & 31;
  const int m = lane & 15, kh = lane >> 4;
  const bf16* r = p + m * ld;
  v8bf lo = *(const v8bf*)(r + kh * 8);
  v8bf hi = *(const v8bf*)(r + 16 + kh * 8);
  v16bf a;
#pragma unroll
  for (int i = 0; i < 8; ++i) { a[i] = lo[i]; a[i + 8] = hi[i]; }
  return a;
}

// B fragment: 32x16 (KxN) from B^T storage (N-major, K contiguous, ld).
DEVFN v16bf load_bt_frag(const bf16* p, int ld) {
  const int lane = threadIdx.x & 31;
  const int n = lane & 15, kh = lane >> 4;
  const bf16* r = p + n * ld + kh * 8;
  v8bf lo = *(const v8bf*)(r);
  v8bf hi = *(const v8bf*)(r + 16);
  v16bf b;
#pragma unroll
  for (int i = 0; i < 8; ++i) { b[i] = lo[i]; b[i + 8] = hi[i]; }
  return b;
}

// Stage a ROWSx32 bf16 tile into LDS; all global loads issue before any store.
template <int ROWS>
DEVFN void load_tile(bf16* lds, const bf16* g, int ldg) {
  constexpr int NCH = ROWS * 4 / 256;    // 16B chunks per thread
  v8bf t[NCH];
#pragma unroll
  for (int i = 0; i < NCH; ++i) {
    const int c = threadIdx.x + i * 256;
    t[i] = *(const v8bf*)(g + (size_t)(c >> 2) * ldg + (c & 3) * 8);
  }
#pragma unroll
  for (int i = 0; i < NCH; ++i) {
    const int c = threadIdx.x + i * 256;
    *(v8bf*)(lds + (c >> 2) * TLD + (c & 3) * 8) = t[i];
  }
}

// Double-buffered K-loop: C[256x128] += A[256xK] * B[Kx128].
// 8 waves in a 4x2 grid, each owning a 64x64 tile (acc[4][4]).
// All 8 fragments (4 A + 4 B) are loaded before the 16 WMMAs so the ds_load
// waits stagger instead of stalling each 4-WMMA cluster.
DEVFN void gemm_loop(const bf16* __restrict__ A, int lda,
                     const bf16* __restrict__ Bt, int ldbt, int Kdim,
                     bf16* sA, bf16* sB, v8f (&acc)[4][4]) {
  const int wid = threadIdx.x >> 5;
  const int wr = wid & 3, wc = wid >> 2;
  int cur = 0;
  load_tile<256>(sA, A, lda);
  load_tile<128>(sB, Bt, ldbt);
  for (int k0 = 0; k0 < Kdim; k0 += 32) {
    __syncthreads();
    const int nxt = cur ^ 1;
    if (k0 + 32 < Kdim) {
      load_tile<256>(sA + nxt * TILE_A, A + k0 + 32, lda);
      load_tile<128>(sB + nxt * TILE_B, Bt + k0 + 32, ldbt);
      if (k0 + 64 < Kdim) {  // L2 hint for the tile after next
        __builtin_prefetch(A + (size_t)threadIdx.x * lda + k0 + 64, 0, 1);
        __builtin_prefetch(Bt + (size_t)(threadIdx.x >> 1) * ldbt + k0 + 64, 0, 1);
      }
    }
    const bf16* a  = sA + cur * TILE_A;
    const bf16* bt = sB + cur * TILE_B;
    v16bf af[4], bfr[4];
#pragma unroll
    for (int mi = 0; mi < 4; ++mi)
      af[mi] = load_a_frag(a + (wr * 64 + mi * 16) * TLD, TLD);
#pragma unroll
    for (int ni = 0; ni < 4; ++ni)
      bfr[ni] = load_bt_frag(bt + (wc * 64 + ni * 16) * TLD, TLD);
#pragma unroll
    for (int ni = 0; ni < 4; ++ni)
#pragma unroll
      for (int mi = 0; mi < 4; ++mi)
        acc[mi][ni] = wmma_bf16(af[mi], bfr[ni], acc[mi][ni]);
    cur = nxt;
  }
}

// ---------------------------------------------------------------------------
// Kernels
// ---------------------------------------------------------------------------
__global__ void cast_f32_bf16_kernel(const float* __restrict__ src,
                                     bf16* __restrict__ dst, int n) {
  const int i4 = (blockIdx.x * blockDim.x + threadIdx.x) * 4;
  if (i4 < n) {
    v4f x = *(const v4f*)(src + i4);
    v4bf y;
#pragma unroll
    for (int j = 0; j < 4; ++j) y[j] = (bf16)x[j];
    *(v4bf*)(dst + i4) = y;
  }
}

// proj = kq @ W^T + bias, stored TRANSPOSED bf16: outT[b][v][n].
__global__ void __launch_bounds__(256, 1)
proj_gemm_T_kernel(const bf16* __restrict__ inB, const bf16* __restrict__ W,
                   const float* __restrict__ bias, bf16* __restrict__ outT) {
  __shared__ __align__(32) bf16 sA[2 * TILE_A];
  __shared__ __align__(32) bf16 sB[2 * TILE_B];
  const int colT = blockIdx.x * 128;             // v
  const int rowT = blockIdx.y * 256;             // b*N + n
  const int b = rowT / kN;
  const int n0 = rowT % kN;
  v8f acc[4][4] = {{{0}}};
  gemm_loop(inB + (size_t)rowT * kV + (kV - kKQ), kV,
            W + (size_t)colT * kKQ, kKQ, kKQ, sA, sB, acc);

  const int wid = threadIdx.x >> 5, wr = wid & 3, wc = wid >> 2;
  const int lane = threadIdx.x & 31, ln = lane & 15, mh = lane >> 4;
#pragma unroll
  for (int ni = 0; ni < 4; ++ni) {
    const int col = colT + wc * 64 + ni * 16 + ln;  // v
    const float bb = bias[col];
#pragma unroll
    for (int mi = 0; mi < 4; ++mi) {
      const int nrow = n0 + wr * 64 + mi * 16 + mh * 8;  // n
      bf16* p = outT + ((size_t)b * kV + col) * kN + nrow;
#pragma unroll
      for (int r = 0; r < 8; ++r) p[r] = (bf16)(acc[mi][ni][r] + bb);
    }
  }
}

// v = input @ Wv^T + bv, row-major bf16 output vB[b][n][v].
__global__ void __launch_bounds__(256, 1)
v_gemm_kernel(const bf16* __restrict__ inB, const bf16* __restrict__ WvB,
              const float* __restrict__ bv, bf16* __restrict__ vB) {
  __shared__ __align__(32) bf16 sA[2 * TILE_A];
  __shared__ __align__(32) bf16 sB[2 * TILE_B];
  const int colT = blockIdx.x * 128;
  const int rowT = blockIdx.y * 256;
  v8f acc[4][4] = {{{0}}};
  gemm_loop(inB + (size_t)rowT * kV, kV, WvB + (size_t)colT * kV, kV, kV, sA, sB, acc);

  const int wid = threadIdx.x >> 5, wr = wid & 3, wc = wid >> 2;
  const int lane = threadIdx.x & 31, ln = lane & 15, mh = lane >> 4;
#pragma unroll
  for (int ni = 0; ni < 4; ++ni) {
    const int col = colT + wc * 64 + ni * 16 + ln;
    const float bb = bv[col];
#pragma unroll
    for (int mi = 0; mi < 4; ++mi) {
      const int row = rowT + wr * 64 + mi * 16 + mh * 8;
      bf16* p = vB + (size_t)row * kV + col;
#pragma unroll
      for (int r = 0; r < 8; ++r) p[(size_t)r * kV] = (bf16)(acc[mi][ni][r] + bb);
    }
  }
}

// attn[b][v][w] = sum_n kT[b][v][n] * qT[b][w][n]  (f32 logits)
__global__ void __launch_bounds__(256, 1)
attn_gemm_kernel(const bf16* __restrict__ kT, const bf16* __restrict__ qT,
                 float* __restrict__ attnL) {
  __shared__ __align__(32) bf16 sA[2 * TILE_A];
  __shared__ __align__(32) bf16 sB[2 * TILE_B];
  const int colT = blockIdx.x * 128;     // w
  const int rowT = blockIdx.y * 256;     // v
  const int b = blockIdx.z;
  v8f acc[4][4] = {{{0}}};
  gemm_loop(kT + ((size_t)b * kV + rowT) * kN, kN,
            qT + ((size_t)b * kV + colT) * kN, kN, kN, sA, sB, acc);

  const int wid = threadIdx.x >> 5, wr = wid & 3, wc = wid >> 2;
  const int lane = threadIdx.x & 31, ln = lane & 15, mh = lane >> 4;
#pragma unroll
  for (int ni = 0; ni < 4; ++ni) {
    const int col = colT + wc * 64 + ni * 16 + ln;
#pragma unroll
    for (int mi = 0; mi < 4; ++mi) {
      const int row = rowT + wr * 64 + mi * 16 + mh * 8;
      float* p = attnL + ((size_t)b * kV + row) * kV + col;
#pragma unroll
      for (int r = 0; r < 8; ++r) p[(size_t)r * kV] = acc[mi][ni][r];
    }
  }
}

// softmax over w for each (b, v) row; store TRANSPOSED bf16: attnPT[b][w][v]
__global__ void softmax_T_kernel(const float* __restrict__ attnL,
                                 bf16* __restrict__ attnPT) {
  const int bv = blockIdx.x;            // b*kV + v
  const int b = bv >> 10, vr = bv & (kV - 1);
  const float* row = attnL + (size_t)bv * kV;
  __shared__ float red[256];
  const int t = threadIdx.x;
  v4f x = *(const v4f*)(row + t * 4);
  float m = fmaxf(fmaxf(x[0], x[1]), fmaxf(x[2], x[3]));
  red[t] = m; __syncthreads();
  for (int s = 128; s > 0; s >>= 1) {
    if (t < s) red[t] = fmaxf(red[t], red[t + s]);
    __syncthreads();
  }
  m = red[0]; __syncthreads();
  float e[4], sum = 0.f;
#pragma unroll
  for (int j = 0; j < 4; ++j) { e[j] = __expf(x[j] - m); sum += e[j]; }
  red[t] = sum; __syncthreads();
  for (int s = 128; s > 0; s >>= 1) {
    if (t < s) red[t] += red[t + s];
    __syncthreads();
  }
  const float inv = 1.f / red[0];
  bf16* dst = attnPT + (size_t)b * kV * kV + vr;   // [w][vr]
#pragma unroll
  for (int j = 0; j < 4; ++j)
    dst[(size_t)(t * 4 + j) * kV] = (bf16)(e[j] * inv);
}

// out[b][n][w] = gamma * (sum_v vB[b][n][v] * attnP[b][v][w]) + input[b][n][w]
__global__ void __launch_bounds__(256, 1)
out_gemm_kernel(const bf16* __restrict__ vB, const bf16* __restrict__ attnPT,
                const float* __restrict__ input, const float* __restrict__ gamma,
                float* __restrict__ out) {
  __shared__ __align__(32) bf16 sA[2 * TILE_A];
  __shared__ __align__(32) bf16 sB[2 * TILE_B];
  const int colT = blockIdx.x * 128;     // w
  const int rowT = blockIdx.y * 256;     // n
  const int b = blockIdx.z;
  v8f acc[4][4] = {{{0}}};
  gemm_loop(vB + ((size_t)b * kN + rowT) * kV, kV,
            attnPT + ((size_t)b * kV + colT) * kV, kV, kV, sA, sB, acc);

  const float g = gamma[0];
  const int wid = threadIdx.x >> 5, wr = wid & 3, wc = wid >> 2;
  const int lane = threadIdx.x & 31, ln = lane & 15, mh = lane >> 4;
#pragma unroll
  for (int ni = 0; ni < 4; ++ni) {
    const int col = colT + wc * 64 + ni * 16 + ln;
#pragma unroll
    for (int mi = 0; mi < 4; ++mi) {
      const int row = rowT + wr * 64 + mi * 16 + mh * 8;
      const size_t base = ((size_t)b * kN + row) * kV + col;
#pragma unroll
      for (int r = 0; r < 8; ++r)
        out[base + (size_t)r * kV] = g * acc[mi][ni][r] + input[base + (size_t)r * kV];
    }
  }
}

// ---------------------------------------------------------------------------
// Host launch
// ---------------------------------------------------------------------------
extern "C" void kernel_launch(void* const* d_in, const int* in_sizes, int n_in,
                              void* d_out, int out_size, void* d_ws, size_t ws_size,
                              hipStream_t stream) {
  const float* input = (const float*)d_in[0];
  const float* Wq    = (const float*)d_in[1];
  const float* bq    = (const float*)d_in[2];
  const float* Wk    = (const float*)d_in[3];
  const float* bk    = (const float*)d_in[4];
  const float* Wv    = (const float*)d_in[5];
  const float* bv    = (const float*)d_in[6];
  const float* gamma = (const float*)d_in[7];
  float* out = (float*)d_out;

  char* ws = (char*)d_ws;
  size_t off = 0;
  auto alloc = [&](size_t bytes) -> void* {
    void* p = ws + off;
    off += (bytes + 255) & ~(size_t)255;
    return p;
  };
  const size_t nInput = (size_t)kB * kN * kV;
  bf16*  inB    = (bf16*)alloc(nInput * 2);
  bf16*  qT     = (bf16*)alloc(nInput * 2);                // [b][v][n]
  bf16*  kTb    = (bf16*)alloc(nInput * 2);                // [b][v][n]
  bf16*  vBuf   = (bf16*)alloc(nInput * 2);                // [b][n][v]
  float* attnL  = (float*)alloc((size_t)kB * kV * kV * 4);
  bf16*  attnPT = (bf16*)alloc((size_t)kB * kV * kV * 2);  // [b][w][v]
  bf16*  WqB    = (bf16*)alloc((size_t)kV * kKQ * 2);
  bf16*  WkB    = (bf16*)alloc((size_t)kV * kKQ * 2);
  bf16*  WvB    = (bf16*)alloc((size_t)kV * kV * 2);
  (void)ws_size; (void)in_sizes; (void)n_in; (void)out_size;

  // 1) casts to bf16
  cast_f32_bf16_kernel<<<dim3((unsigned)(nInput / 4 / 256)), 256, 0, stream>>>(
      input, inB, (int)nInput);
  cast_f32_bf16_kernel<<<dim3(kV * kKQ / 4 / 256), 256, 0, stream>>>(Wq, WqB, kV * kKQ);
  cast_f32_bf16_kernel<<<dim3(kV * kKQ / 4 / 256), 256, 0, stream>>>(Wk, WkB, kV * kKQ);
  cast_f32_bf16_kernel<<<dim3(kV * kV / 4 / 256), 256, 0, stream>>>(Wv, WvB, kV * kV);

  // 2) q, k, v projections (rows = 32768, macro rows = 256 -> 128 blocks)
  proj_gemm_T_kernel<<<dim3(kV / 128, kB * kN / 256), 256, 0, stream>>>(
      inB, WqB, bq, qT);
  proj_gemm_T_kernel<<<dim3(kV / 128, kB * kN / 256), 256, 0, stream>>>(
      inB, WkB, bk, kTb);
  v_gemm_kernel<<<dim3(kV / 128, kB * kN / 256), 256, 0, stream>>>(inB, WvB, bv, vBuf);

  // 3) attention logits: per-batch 1024x1024, K=4096
  attn_gemm_kernel<<<dim3(kV / 128, kV / 256, kB), 256, 0, stream>>>(kTb, qT, attnL);

  // 4) softmax over last axis, store transposed bf16
  softmax_T_kernel<<<dim3(kB * kV), 256, 0, stream>>>(attnL, attnPT);

  // 5) out = gamma * (v @ attn) + input
  out_gemm_kernel<<<dim3(kV / 128, kN / 256, kB), 256, 0, stream>>>(
      vBuf, attnPT, input, gamma, out);
}
